// TemporalMambaWrapper_22497038696547
// MI455X (gfx1250) — matrix operand order repaired and checked by reading, not compiled, not run on hardware
//
#include <hip/hip_runtime.h>
#include <hip/hip_bf16.h>

typedef _Float16 h16;
typedef __attribute__((ext_vector_type(16))) _Float16 v16h;
typedef __attribute__((ext_vector_type(8)))  float    v8f;
typedef __attribute__((ext_vector_type(4)))  uint32_t v4u;
typedef __attribute__((ext_vector_type(8)))  uint32_t v8u;

#define BB      8
#define CC      128
#define TT      64
#define NSEQ    256
#define DSTATE  4
#define DINNER  128
#define DTRANK  8

// f16 weight layout inside d_ws (element offsets)
#define WS_INW   0              // (256,128)  -> 32768
#define WS_OUTW  32768          // (128,128)  -> 16384
#define WS_XPW   49152          // (16,128)   -> 2048
#define WS_DTW   51200          // (128,8)    -> 1024
#define WS_TOTAL 52224

// LDS padding
#define PADH 136   // f16 rows (128 + 8)
#define PADF 129   // f32 rows (128 + 1)
#define PADW 130   // TDM-padded weight rows: 256B + 4B pad per row

__device__ __forceinline__ float fast_silu(float a) {
    // a * rcp(1 + exp(-a)) : v_exp_f32 + v_rcp_f32, no IEEE div chain
    return a * __builtin_amdgcn_rcpf(1.0f + __expf(-a));
}

// ---------------------------------------------------------------------------
// Pre-kernel: convert all weight matrices f32 -> f16 into d_ws.
// ---------------------------------------------------------------------------
__global__ void cvt_weights_kernel(const float* __restrict__ in_w,
                                   const float* __restrict__ out_w,
                                   const float* __restrict__ xp_w,
                                   const float* __restrict__ dt_w,
                                   h16* __restrict__ w16) {
    const int i = blockIdx.x * 256 + threadIdx.x;
    if (i < WS_OUTW)        w16[i] = (h16)in_w[i];
    else if (i < WS_XPW)    w16[i] = (h16)out_w[i - WS_OUTW];
    else if (i < WS_DTW)    w16[i] = (h16)xp_w[i - WS_XPW];
    else if (i < WS_TOTAL)  w16[i] = (h16)dt_w[i - WS_DTW];
}

// ---------------------------------------------------------------------------
// Fused Mamba block: one workgroup (4 wave32s) per (batch, node) sequence.
// ---------------------------------------------------------------------------
__launch_bounds__(128)
__global__ void mamba_fused_kernel(
    const float* __restrict__ x,         // (B, C, T, N)
    const float* __restrict__ ln_g,      // (C)
    const float* __restrict__ ln_b,      // (C)
    const float* __restrict__ conv_w,    // (3, 1, DINNER)
    const float* __restrict__ conv_b,    // (DINNER)
    const float* __restrict__ dt_b,      // (DINNER)
    const float* __restrict__ A_log,     // (DINNER, DSTATE)
    const float* __restrict__ Dp,        // (DINNER)
    const h16*   __restrict__ w16,       // converted weights in d_ws
    float* __restrict__ out)             // (B, C, T, N)
{
    __shared__ h16   sA  [TT][PADH];   // LN'd input (A operand); later gated y
    __shared__ float sXPD[TT][PADF];   // xc pre-conv, then reused for dt
    __shared__ float sXC [TT][PADF];   // xc post conv+silu
    __shared__ float sSZ [TT][PADF];   // silu(z)
    __shared__ float sBC [TT][8];      // Bs(0..3), Cs(4..7)
    __shared__ float sDI [TT][8];      // dt-rank input (x_dbl cols 0..7)
    __shared__ h16   sW  [DINNER][PADW]; // out_proj weights, TDM-staged

    const int tid  = threadIdx.x;
    const int lane = tid & 31;
    const int wv   = tid >> 5;       // wave id 0..3
    const int lrow = lane & 15;      // within-tile row/col
    const int lhi  = lane >> 4;      // lane group 0/1

    const int seq  = blockIdx.x;     // 0..2047
    const int bIdx = seq >> 8;
    const int nIdx = seq & 255;

    const h16* in_w16 = w16 + WS_INW;
    const h16* ot_w16 = w16 + WS_OUTW;
    const h16* xp_w16 = w16 + WS_XPW;
    const h16* dt_w16 = w16 + WS_DTW;

    __builtin_prefetch(in_w16 + lane * 128, 0, 1);   // global_prefetch_b8
    __builtin_prefetch(xp_w16, 0, 1);

    // ---- Issue TDM: DMA out_proj f16 weights (128x128) global -> LDS ------
    // D# per CDNA5 ISA: 2D tensor, data_size=2B,
    // pad_enable with interval=64 DWORDs (one row), amount=1 DWORD, so LDS
    // rows land at stride 260B (matches sW[128][130]).
    if (wv == 0) {
        const uint64_t ga  = (uint64_t)(uintptr_t)ot_w16;
        const uint32_t ldsa =
            __builtin_amdgcn_readfirstlane((uint32_t)(uintptr_t)&sW[0][0]);
        const uint32_t galo = __builtin_amdgcn_readfirstlane((uint32_t)ga);
        const uint32_t gahi = __builtin_amdgcn_readfirstlane(
            (uint32_t)((ga >> 32) & 0x01FFFFFFu) | (2u << 30)); // type=2
        v4u g0;
        g0[0] = 1u;        // count=1, user mode
        g0[1] = ldsa;      // lds_addr
        g0[2] = galo;      // global_addr[31:0]
        g0[3] = gahi;      // global_addr[56:32] | type
        v8u g1;
        g1[0] = (1u << 16) | (1u << 20) | (5u << 22) | (0u << 25);
                            // data_size=2B, pad_enable, interval=64DW, amount=1DW
        g1[1] = (128u << 16);   // tensor_dim0[15:0]=128
        g1[2] = (128u << 16);   // tensor_dim0 hi=0 | tensor_dim1[15:0]=128
        g1[3] = (128u << 16);   // tensor_dim1 hi=0 | tile_dim0=128
        g1[4] = 128u;           // tile_dim1=128, tile_dim2=0
        g1[5] = 128u;           // tensor_dim0_stride[31:0]=128
        g1[6] = 0u;
        g1[7] = 0u;
        asm volatile("tensor_load_to_lds %0, %1" : : "s"(g0), "s"(g1) : "memory");
    }

    // ---------------- Stage A: gather + LayerNorm over C -> sA (f16) -------
    {
        const int c0 = lane, c1 = lane + 32, c2 = lane + 64, c3 = lane + 96;
        const float g0 = ln_g[c0], g1 = ln_g[c1], g2 = ln_g[c2], g3 = ln_g[c3];
        const float e0 = ln_b[c0], e1 = ln_b[c1], e2 = ln_b[c2], e3 = ln_b[c3];
        const long base = (long)bIdx * CC * TT * NSEQ + nIdx;
        for (int i = 0; i < 16; ++i) {
            const int t = wv * 16 + i;
            float v0 = x[base + ((long)(c0 * TT + t)) * NSEQ];
            float v1 = x[base + ((long)(c1 * TT + t)) * NSEQ];
            float v2 = x[base + ((long)(c2 * TT + t)) * NSEQ];
            float v3 = x[base + ((long)(c3 * TT + t)) * NSEQ];
            float s = v0 + v1 + v2 + v3;
            float q = v0 * v0 + v1 * v1 + v2 * v2 + v3 * v3;
            #pragma unroll
            for (int off = 16; off >= 1; off >>= 1) {
                s += __shfl_xor(s, off, 32);
                q += __shfl_xor(q, off, 32);
            }
            const float mean = s * (1.0f / 128.0f);
            const float var  = q * (1.0f / 128.0f) - mean * mean;
            const float rstd = rsqrtf(var + 1e-5f);
            sA[t][c0] = (h16)((v0 - mean) * rstd * g0 + e0);
            sA[t][c1] = (h16)((v1 - mean) * rstd * g1 + e1);
            sA[t][c2] = (h16)((v2 - mean) * rstd * g2 + e2);
            sA[t][c3] = (h16)((v3 - mean) * rstd * g3 + e3);
        }
    }
    __syncthreads();

    // ---------------- Stage B: in_proj GEMM (64x128)@(128x256) -------------
    // Split into xc half (cols 0..127) and z half (cols 128..255) so each
    // writeback is branch-free (nj is wave-uniform).
    #pragma unroll
    for (int half = 0; half < 2; ++half) {
        for (int q = wv; q < 32; q += 4) {
            const int mi = q >> 3;              // 0..3
            const int nj = half * 8 + (q & 7);  // 0..7 or 8..15
            v8f acc = {};
            const int m = mi * 16 + lrow;
            const int n = nj * 16 + lrow;
            #pragma unroll
            for (int kb = 0; kb < 4; ++kb) {
                v16h af, bf;
                const int ka = kb * 32 + lhi * 16;
                #pragma unroll
                for (int v = 0; v < 8; ++v) {
                    const int ko = 2 * v;
                    af[2 * v]     = sA[m][ka + ko];
                    af[2 * v + 1] = sA[m][ka + ko + 1];
                }
                const h16* wr = in_w16 + n * CC + kb * 32 + lhi * 16;
                #pragma unroll
                for (int v = 0; v < 8; ++v) {
                    bf[2 * v]     = wr[2 * v];
                    bf[2 * v + 1] = wr[2 * v + 1];
                }
                acc = __builtin_amdgcn_wmma_f32_16x16x32_f16(
                    false, af, false, bf, (short)0, acc, false, false);
            }
            #pragma unroll
            for (int v = 0; v < 8; ++v) {
                const int mt = mi * 16 + v + 8 * lhi;
                const int jn = nj * 16 + lrow;
                if (half == 0) sXPD[mt][jn] = acc[v];                 // xc pre-conv
                else           sSZ[mt][jn - DINNER] = fast_silu(acc[v]); // silu(z)
            }
        }
    }
    __syncthreads();

    // ---------------- Stage C: causal depthwise conv (K=3) + SiLU ----------
    {
        const int d = tid;
        const float c0 = conv_w[0 * DINNER + d];
        const float c1 = conv_w[1 * DINNER + d];
        const float c2 = conv_w[2 * DINNER + d];
        const float cb = conv_b[d];
        for (int t = 0; t < TT; ++t) {
            const float xm2 = (t >= 2) ? sXPD[t - 2][d] : 0.0f;
            const float xm1 = (t >= 1) ? sXPD[t - 1][d] : 0.0f;
            const float x0  = sXPD[t][d];
            const float a   = c0 * xm2 + c1 * xm1 + c2 * x0 + cb;
            sXC[t][d] = fast_silu(a);
        }
    }
    __syncthreads();

    // ---------------- Stage D: x_proj GEMM (64x128)@(128x16) ---------------
    {
        const int mi = wv;
        v8f acc = {};
        const int m = mi * 16 + lrow;
        const int n = lrow;
        #pragma unroll
        for (int kb = 0; kb < 4; ++kb) {
            v16h af, bf;
            const int ka = kb * 32 + lhi * 16;
            #pragma unroll
            for (int v = 0; v < 8; ++v) {
                const int ko = 2 * v;
                af[2 * v]     = (h16)sXC[m][ka + ko];
                af[2 * v + 1] = (h16)sXC[m][ka + ko + 1];
            }
            const h16* wr = xp_w16 + n * DINNER + kb * 32 + lhi * 16;
            #pragma unroll
            for (int v = 0; v < 8; ++v) {
                bf[2 * v]     = wr[2 * v];
                bf[2 * v + 1] = wr[2 * v + 1];
            }
            acc = __builtin_amdgcn_wmma_f32_16x16x32_f16(
                false, af, false, bf, (short)0, acc, false, false);
        }
        #pragma unroll
        for (int v = 0; v < 8; ++v) {
            const int mt = mi * 16 + v + 8 * lhi;
            const int jn = lrow;
            if (jn < DTRANK) sDI[mt][jn] = acc[v];
            else             sBC[mt][jn - DTRANK] = acc[v];
        }
    }
    __syncthreads();

    // ---------------- Stage E: dt_proj GEMM (64x8)@(8x128), K padded -------
    // (writes dt into sXPD, which is dead after the conv stage)
    for (int q = wv; q < 32; q += 4) {
        const int mi = q >> 3;
        const int nj = q & 7;
        v8f acc = {};
        const int m = mi * 16 + lrow;
        const int n = nj * 16 + lrow;
        v16h af, bf;
        const int ka = lhi * 16;
        #pragma unroll
        for (int v = 0; v < 8; ++v) {
            const int ko = 2 * v;
            const int k0 = ka + ko;
            af[2 * v]     = (k0     < DTRANK) ? (h16)sDI[m][k0]     : (h16)0.0f;
            af[2 * v + 1] = (k0 + 1 < DTRANK) ? (h16)sDI[m][k0 + 1] : (h16)0.0f;
        }
        const int kbp = lhi * 16;
        #pragma unroll
        for (int v = 0; v < 8; ++v) {
            const int k0 = kbp + 2 * v;
            bf[2 * v]     = (k0     < DTRANK) ? dt_w16[n * DTRANK + k0]     : (h16)0.0f;
            bf[2 * v + 1] = (k0 + 1 < DTRANK) ? dt_w16[n * DTRANK + k0 + 1] : (h16)0.0f;
        }
        acc = __builtin_amdgcn_wmma_f32_16x16x32_f16(
            false, af, false, bf, (short)0, acc, false, false);
        #pragma unroll
        for (int v = 0; v < 8; ++v) {
            const int mt = mi * 16 + v + 8 * lhi;
            const int jn = nj * 16 + lrow;
            float val = acc[v] + dt_b[jn];
            val = (val > 20.0f) ? val : log1pf(__expf(val));  // softplus
            sXPD[mt][jn] = val;                               // dt
        }
    }
    __syncthreads();

    // ---------------- Stage F: selective scan over T (serial), gate --------
    {
        const int d = tid;
        const float a0 = -__expf(A_log[d * DSTATE + 0]);
        const float a1 = -__expf(A_log[d * DSTATE + 1]);
        const float a2 = -__expf(A_log[d * DSTATE + 2]);
        const float a3 = -__expf(A_log[d * DSTATE + 3]);
        const float Dd = Dp[d];
        float h0 = 0.f, h1 = 0.f, h2 = 0.f, h3 = 0.f;
        for (int t = 0; t < TT; ++t) {
            const float dtv = sXPD[t][d];
            const float xcv = sXC[t][d];
            const float dbx = dtv * xcv;
            const float b0 = sBC[t][0], b1 = sBC[t][1], b2 = sBC[t][2], b3 = sBC[t][3];
            const float g0 = sBC[t][4], g1 = sBC[t][5], g2 = sBC[t][6], g3 = sBC[t][7];
            h0 = h0 * __expf(dtv * a0) + dbx * b0;
            h1 = h1 * __expf(dtv * a1) + dbx * b1;
            h2 = h2 * __expf(dtv * a2) + dbx * b2;
            h3 = h3 * __expf(dtv * a3) + dbx * b3;
            float y = h0 * g0 + h1 * g1 + h2 * g2 + h3 * g3;
            y += xcv * Dd;
            y *= sSZ[t][d];
            sA[t][d] = (h16)y;          // A operand for out_proj
        }
    }
    // Ensure the TDM transfer has landed before any wave touches sW.
    if (wv == 0) __builtin_amdgcn_s_wait_tensorcnt(0);
    __syncthreads();

    // ---------------- Stage G: out_proj GEMM (64x128)@(128x128) ------------
    // B operand comes from the TDM-staged LDS copy of out_proj_w (f16).
    {
        const long obase = (long)bIdx * CC * TT * NSEQ + nIdx;
        for (int q = wv; q < 32; q += 4) {
            const int mi = q >> 3;
            const int nj = q & 7;
            v8f acc = {};
            const int m = mi * 16 + lrow;
            const int n = nj * 16 + lrow;
            #pragma unroll
            for (int kb = 0; kb < 4; ++kb) {
                v16h af, bf;
                const int ka = kb * 32 + lhi * 16;
                #pragma unroll
                for (int v = 0; v < 8; ++v) {
                    const int ko = 2 * v;
                    af[2 * v]     = sA[m][ka + ko];
                    af[2 * v + 1] = sA[m][ka + ko + 1];
                }
                const int kbp = kb * 32 + lhi * 16;
                #pragma unroll
                for (int v = 0; v < 8; ++v) {
                    bf[2 * v]     = sW[n][kbp + 2 * v];
                    bf[2 * v + 1] = sW[n][kbp + 2 * v + 1];
                }
                acc = __builtin_amdgcn_wmma_f32_16x16x32_f16(
                    false, af, false, bf, (short)0, acc, false, false);
            }
            #pragma unroll
            for (int v = 0; v < 8; ++v) {
                const int t = mi * 16 + v + 8 * lhi;
                const int c = nj * 16 + lrow;
                out[obase + ((long)(c * TT + t)) * NSEQ] = acc[v];
            }
        }
    }
}

extern "C" void kernel_launch(void* const* d_in, const int* in_sizes, int n_in,
                              void* d_out, int out_size, void* d_ws, size_t ws_size,
                              hipStream_t stream) {
    const float* x      = (const float*)d_in[0];
    const float* ln_g   = (const float*)d_in[1];
    const float* ln_b   = (const float*)d_in[2];
    const float* in_w   = (const float*)d_in[3];
    const float* conv_w = (const float*)d_in[4];
    const float* conv_b = (const float*)d_in[5];
    const float* xp_w   = (const float*)d_in[6];
    const float* dt_w   = (const float*)d_in[7];
    const float* dt_b   = (const float*)d_in[8];
    const float* A_log  = (const float*)d_in[9];
    const float* Dp     = (const float*)d_in[10];
    const float* out_w  = (const float*)d_in[11];
    float* out = (float*)d_out;
    h16* w16 = (h16*)d_ws;

    // 1) convert weights f32 -> f16 into workspace
    cvt_weights_kernel<<<(WS_TOTAL + 255) / 256, 256, 0, stream>>>(
        in_w, out_w, xp_w, dt_w, w16);

    // 2) fused mamba: one WG (4 wave32s) per sequence
    dim3 grid(BB * NSEQ);
    dim3 block(128);
    mamba_fused_kernel<<<grid, block, 0, stream>>>(
        x, ln_g, ln_b, conv_w, conv_b, dt_b, A_log, Dp, w16, out);
}
